// AttnPlus_55937654063649
// MI455X (gfx1250) — matrix operand order repaired and checked by reading, not compiled, not run on hardware
//
#include <hip/hip_runtime.h>
#include <math.h>

typedef __attribute__((ext_vector_type(16))) _Float16 v16h;
typedef __attribute__((ext_vector_type(8)))  _Float16 v8h;
typedef __attribute__((ext_vector_type(8)))  float    v8f;

#define B_    4
#define N_    2048
#define DIM_  1024
#define HEAD_ 16
#define HD_   64
#define LN_EPS 1e-5f

#if defined(__HIP_DEVICE_COMPILE__) && defined(__gfx1250__) && \
    __has_builtin(__builtin_amdgcn_tensor_load_to_lds) &&      \
    __has_builtin(__builtin_amdgcn_s_wait_tensorcnt)
#define USE_TDM 1
#else
#define USE_TDM 0
#endif

// ---------------- Kernel 1: LayerNorm -> f16 ----------------
__global__ __launch_bounds__(256) void ln_to_h(const float* __restrict__ x,
                                               const float* __restrict__ w,
                                               const float* __restrict__ b,
                                               _Float16* __restrict__ xn) {
    __shared__ float red[256];
    const int row = blockIdx.x;                 // 0..B*N-1
    const int t   = threadIdx.x;
    const float* xr = x + (size_t)row * DIM_;
    float v[4];
    float s = 0.f;
#pragma unroll
    for (int i = 0; i < 4; ++i) { v[i] = xr[t + 256 * i]; s += v[i]; }
    red[t] = s; __syncthreads();
    for (int o = 128; o > 0; o >>= 1) { if (t < o) red[t] += red[t + o]; __syncthreads(); }
    const float mu = red[0] * (1.0f / DIM_);
    __syncthreads();
    float sv = 0.f;
#pragma unroll
    for (int i = 0; i < 4; ++i) { float c = v[i] - mu; sv += c * c; }
    red[t] = sv; __syncthreads();
    for (int o = 128; o > 0; o >>= 1) { if (t < o) red[t] += red[t + o]; __syncthreads(); }
    const float rstd = rsqrtf(red[0] * (1.0f / DIM_) + LN_EPS);
    _Float16* xo = xn + (size_t)row * DIM_;
#pragma unroll
    for (int i = 0; i < 4; ++i) {
        const int c = t + 256 * i;
        xo[c] = (_Float16)((v[i] - mu) * rstd * w[c] + b[c]);
    }
}

// ---------------- Kernel 2: Wqk f32 -> f16 ----------------
__global__ __launch_bounds__(256) void wqk_to_h(const float* __restrict__ Wqk,
                                                _Float16* __restrict__ Wh, int n) {
    const int i = blockIdx.x * 256 + threadIdx.x;
    if (i < n) Wh[i] = (_Float16)Wqk[i];
}

// ---------------- TDM descriptor helper: 64 rows x 32 halves tile -> LDS ----------------
#if USE_TDM
typedef __attribute__((ext_vector_type(4))) unsigned int u32x4;
typedef __attribute__((ext_vector_type(8))) int          i32x8;
typedef __attribute__((ext_vector_type(4))) int          i32x4;

__device__ inline void tdm_load_tile(const _Float16* gsrc, unsigned lds_byte_off) {
    const unsigned long long ga = (unsigned long long)(uintptr_t)gsrc;
    u32x4 g0;
    g0[0] = 1u;                                   // count=1 (valid user D#)
    g0[1] = lds_byte_off;                         // lds_addr (bytes)
    g0[2] = (unsigned)(ga & 0xFFFFFFFFu);         // global_addr[31:0]
    g0[3] = (unsigned)((ga >> 32) & 0x1FFFFFFu)   // global_addr[56:32]
          | 0x80000000u;                          // type=2 ("image")
    i32x8 g1;
    g1[0] = (1 << 16)                             // data_size = 2 bytes
          | (1 << 20)                             // pad_enable (LDS dest padding)
          | (3 << 22)                             // pad_interval: 16 DWORDs (64B row)
          | (3 << 25);                            // pad_amount: 4 DWORDs (16B pad)
    g1[1] = (DIM_ & 0xFFFF) << 16;                // tensor_dim0[15:0]  (bits 63:48)
    g1[2] = ((DIM_ >> 16) & 0xFFFF)               // tensor_dim0[31:16] (bits 79:64)
          | (((2 * DIM_) & 0xFFFF) << 16);        // tensor_dim1[15:0]  (bits 95:80)
    g1[3] = (((2 * DIM_) >> 16) & 0xFFFF)         // tensor_dim1[31:16] (bits 111:96)
          | (32 << 16);                           // tile_dim0 = 32 halves (bits 127:112)
    g1[4] = 64;                                   // tile_dim1 = 64 rows (bits 143:128)
    g1[5] = DIM_;                                 // tensor_dim0_stride[31:0]
    g1[6] = 0;                                    // stride hi bits / dim1_stride lo
    g1[7] = 0;
    const i32x4 gz = {0, 0, 0, 0};                // groups 2/3: tile_dim2..4 = 0 (unused)
#if __clang_major__ >= 23
    const i32x8 gz8 = {0, 0, 0, 0, 0, 0, 0, 0};
    __builtin_amdgcn_tensor_load_to_lds(g0, g1, gz, gz, gz8, 0);
#else
    __builtin_amdgcn_tensor_load_to_lds(g0, g1, gz, gz, 0);
#endif
}
#endif

// ---------------- Kernel 3: QK projection GEMM ----------------
// C[m, e] = sum_d xn[m,d] * Wqk[e,d];  M=8192, Nout=2048, K=1024
// Block tile: 256(M) x 64(N); B k-slice staged once per block in LDS (TDM when available).
// Wave tile: 32(M) x 64(N) -> 8 WMMAs per k-step sharing one B fragment set.
#define BSTRIDE 40   // padded LDS row stride (halves): 64B data + 16B pad per row
__global__ __launch_bounds__(256) void qk_gemm(const _Float16* __restrict__ xn,
                                               const _Float16* __restrict__ Wh,
                                               _Float16* __restrict__ qh,
                                               _Float16* __restrict__ kh) {
    __shared__ _Float16 Bs[64 * BSTRIDE];       // 5 KB
    const int lane = threadIdx.x & 31;
    const int wave = threadIdx.x >> 5;
    const int bm   = blockIdx.x >> 5;           // 32 M-blocks of 256 rows
    const int bn   = blockIdx.x & 31;           // 32 N-blocks of 64 cols
    const int m0   = bm * 256 + wave * 32;
    const int n0   = bn * 64;
    const int col  = lane & 15;
    const int g    = lane >> 4;

#if USE_TDM
    const unsigned ldsBase = (unsigned)(uintptr_t)&Bs[0];
#else
    // cooperative staging: thread t moves 8 halves (16B) of the 64x32 slice
    const int sc = threadIdx.x >> 2;            // staged column 0..63
    const int sk = (threadIdx.x & 3) * 8;       // k offset 0,8,16,24
    const _Float16* bsrc = Wh + (size_t)(n0 + sc) * DIM_ + sk;
#endif

    v8f acc[2][4] = {};
    const _Float16* arow[2] = { xn + (size_t)(m0 + col) * DIM_,
                                xn + (size_t)(m0 + 16 + col) * DIM_ };

    for (int k0 = 0; k0 < DIM_; k0 += 32) {
        __syncthreads();                        // previous-step LDS reads done
#if USE_TDM
        if (wave == 0) {
            tdm_load_tile(Wh + (size_t)n0 * DIM_ + k0, ldsBase);
            __builtin_amdgcn_s_wait_tensorcnt(0);
        }
#else
        *(v8h*)&Bs[sc * BSTRIDE + sk] = *(const v8h*)(bsrc + k0);
#endif
        __syncthreads();                        // tile visible to all waves
        if (k0 + 32 < DIM_) {                   // gfx1250 global_prefetch_b8
            __builtin_prefetch(arow[0] + k0 + 32, 0, 1);
            __builtin_prefetch(arow[1] + k0 + 32, 0, 1);
#if !USE_TDM
            __builtin_prefetch(bsrc + k0 + 32, 0, 1);
#endif
        }
        v16h a[2];
#pragma unroll
        for (int rt = 0; rt < 2; ++rt)
#pragma unroll
            for (int e = 0; e < 16; ++e) {
                const int k = k0 + 16 * g + e;
                a[rt][e] = arow[rt][k];
            }
#pragma unroll
        for (int nn = 0; nn < 4; ++nn) {
            const _Float16* bp = &Bs[(nn * 16 + col) * BSTRIDE + 16 * g];
            v16h bf;
#pragma unroll
            for (int e = 0; e < 16; ++e) bf[e] = bp[e];
            acc[0][nn] = __builtin_amdgcn_wmma_f32_16x16x32_f16(
                false, a[0], false, bf, (short)0, acc[0][nn], false, false);
            acc[1][nn] = __builtin_amdgcn_wmma_f32_16x16x32_f16(
                false, a[1], false, bf, (short)0, acc[1][nn], false, false);
        }
    }
    // scatter-store into q/k [B,H,N,HD] f16; row m = m0 + rt*16 + r + 8g
#pragma unroll
    for (int nn = 0; nn < 4; ++nn) {
        const int e_out = n0 + nn * 16 + col;
        const int sel = e_out >> 10;            // 0 = q, 1 = k
        const int rem = e_out & 1023;
        const int h   = rem >> 6;
        const int d   = rem & 63;
        _Float16* dst = sel ? kh : qh;
#pragma unroll
        for (int rt = 0; rt < 2; ++rt)
#pragma unroll
            for (int r = 0; r < 8; ++r) {
                const int m  = m0 + rt * 16 + r + 8 * g;
                const int bb = m >> 11;
                const int n  = m & (N_ - 1);
                dst[(((size_t)bb * HEAD_ + h) * N_ + n) * HD_ + d] =
                    (_Float16)acc[rt][nn][r];
            }
    }
}

// ---------------- Kernel 4: streaming attention (flash-style, scalar v) ----------------
// Wave handles 32 query rows (2 tiles); each K fragment feeds 2 WMMAs.
__global__ __launch_bounds__(256) void attn(const _Float16* __restrict__ qh,
                                            const _Float16* __restrict__ kh,
                                            const float* __restrict__ Ain,
                                            const float* __restrict__ wv,
                                            float* __restrict__ out) {
    const int lane = threadIdx.x & 31;
    const int wave = threadIdx.x >> 5;
    const int bh   = blockIdx.x;                 // 0..B*HEAD-1
    const int b    = bh >> 4;
    const int h    = bh & 15;
    const int q0   = blockIdx.y * 256 + wave * 32;
    const int col  = lane & 15;
    const int g    = lane >> 4;
    const float wvv   = wv[0];
    const float scale = 0.03125f;                // DIM^-0.5

    v16h aq[2][2];                               // [row tile][K chunk]
#pragma unroll
    for (int rt = 0; rt < 2; ++rt) {
        const _Float16* qbase = qh + ((size_t)bh * N_ + (q0 + rt * 16 + col)) * HD_;
#pragma unroll
        for (int c = 0; c < 2; ++c)
#pragma unroll
            for (int e = 0; e < 16; ++e) {
                const int k = 32 * c + 16 * g + e;
                aq[rt][c][e] = qbase[k];
            }
    }

    float rmax[2][8], rsum[2][8], racc[2][8];
#pragma unroll
    for (int rt = 0; rt < 2; ++rt)
#pragma unroll
        for (int r = 0; r < 8; ++r) {
            rmax[rt][r] = -INFINITY; rsum[rt][r] = 0.f; racc[rt][r] = 0.f;
        }

    for (int j0 = 0; j0 < N_; j0 += 16) {
        v8f c8[2] = {};
        const _Float16* kb0 = kh + ((size_t)bh * N_ + (j0 + col)) * HD_;
        if (j0 + 16 < N_) __builtin_prefetch(kb0 + 16 * HD_, 0, 1);
#pragma unroll
        for (int c = 0; c < 2; ++c) {
            const _Float16* kb = kb0 + 32 * c + 16 * g;
            v16h bf;
#pragma unroll
            for (int e = 0; e < 16; ++e) bf[e] = kb[e];
            c8[0] = __builtin_amdgcn_wmma_f32_16x16x32_f16(
                false, aq[0][c], false, bf, (short)0, c8[0], false, false);
            c8[1] = __builtin_amdgcn_wmma_f32_16x16x32_f16(
                false, aq[1][c], false, bf, (short)0, c8[1], false, false);
        }
        const float vcol = Ain[((size_t)b * N_ + (j0 + col)) * HEAD_ + h] * wvv;
#pragma unroll
        for (int rt = 0; rt < 2; ++rt)
#pragma unroll
            for (int r = 0; r < 8; ++r) {
                const float s = c8[rt][r] * scale;
                float bm = s;
#pragma unroll
                for (int m = 1; m < 16; m <<= 1) bm = fmaxf(bm, __shfl_xor(bm, m, 32));
                const float nm = fmaxf(rmax[rt][r], bm);
                float p  = __expf(s - nm);
                float pv = p * vcol;
#pragma unroll
                for (int m = 1; m < 16; m <<= 1) {
                    p  += __shfl_xor(p, m, 32);
                    pv += __shfl_xor(pv, m, 32);
                }
                const float alpha = __expf(rmax[rt][r] - nm);
                rsum[rt][r] = rsum[rt][r] * alpha + p;
                racc[rt][r] = racc[rt][r] * alpha + pv;
                rmax[rt][r] = nm;
            }
    }
    if (col == 0) {
#pragma unroll
        for (int rt = 0; rt < 2; ++rt)
#pragma unroll
            for (int r = 0; r < 8; ++r) {
                const int n = q0 + rt * 16 + r + 8 * g;
                const size_t idx = ((size_t)b * N_ + n) * HEAD_ + h;
                out[idx] = Ain[idx] + racc[rt][r] / rsum[rt][r];
            }
    }
}

extern "C" void kernel_launch(void* const* d_in, const int* in_sizes, int n_in,
                              void* d_out, int out_size, void* d_ws, size_t ws_size,
                              hipStream_t stream) {
    const float* x    = (const float*)d_in[0];
    const float* A    = (const float*)d_in[1];
    const float* ln_w = (const float*)d_in[2];
    const float* ln_b = (const float*)d_in[3];
    const float* Wqk  = (const float*)d_in[4];
    const float* wv   = (const float*)d_in[5];
    float* out        = (float*)d_out;

    char* ws = (char*)d_ws;
    size_t off = 0;
    _Float16* xn = (_Float16*)(ws + off); off += (size_t)B_ * N_ * DIM_ * 2;        // 16 MB
    _Float16* Wh = (_Float16*)(ws + off); off += (size_t)2 * DIM_ * DIM_ * 2;       //  4 MB
    _Float16* qh = (_Float16*)(ws + off); off += (size_t)B_ * HEAD_ * N_ * HD_ * 2; // 16 MB
    _Float16* kh = (_Float16*)(ws + off); off += (size_t)B_ * HEAD_ * N_ * HD_ * 2; // 16 MB
    if (ws_size < off) return;

    ln_to_h<<<B_ * N_, 256, 0, stream>>>(x, ln_w, ln_b, xn);
    wqk_to_h<<<(2 * DIM_ * DIM_) / 256, 256, 0, stream>>>(Wqk, Wh, 2 * DIM_ * DIM_);
    qk_gemm<<<(B_ * N_ / 256) * ((2 * DIM_) / 64), 256, 0, stream>>>(xn, Wh, qh, kh);
    attn<<<dim3(B_ * HEAD_, N_ / 256), 256, 0, stream>>>(qh, kh, A, wv, out);
}
